// LiZAttention_41214506172509
// MI455X (gfx1250) — compile-verified, matrix operand order checked by reading, not run on hardware
//
#include <hip/hip_runtime.h>

typedef __bf16 bf16_t;
typedef __bf16 v16bf __attribute__((ext_vector_type(16)));
typedef float  v8f   __attribute__((ext_vector_type(8)));
typedef unsigned int v4u __attribute__((ext_vector_type(4)));
typedef int v4i __attribute__((ext_vector_type(4)));
typedef int v8i __attribute__((ext_vector_type(8)));

#if defined(__has_builtin)
#if __has_builtin(__builtin_amdgcn_tensor_load_to_lds) && __has_builtin(__builtin_amdgcn_s_wait_tensorcnt)
#define HAVE_TDM 1
#endif
#endif
#ifndef HAVE_TDM
#define HAVE_TDM 0
#endif

// ---------------------------------------------------------------- helpers
__device__ __forceinline__ int kmap(int p, int hsel) {
  // 16-bit A/B matrix K mapping (CDNA5 ISA 7.12.2): VGPR pair p -> K index
  int kk = (p < 4) ? (2 * p) : (16 + 2 * (p - 4));
  return kk + (hsel ? 8 : 0);
}
__device__ __forceinline__ float red16_max(float v) {
  v = fmaxf(v, __shfl_xor(v, 1, 32));
  v = fmaxf(v, __shfl_xor(v, 2, 32));
  v = fmaxf(v, __shfl_xor(v, 4, 32));
  v = fmaxf(v, __shfl_xor(v, 8, 32));
  return v;
}
__device__ __forceinline__ float red16_sum(float v) {
  v += __shfl_xor(v, 1, 32);
  v += __shfl_xor(v, 2, 32);
  v += __shfl_xor(v, 4, 32);
  v += __shfl_xor(v, 8, 32);
  return v;
}

#if HAVE_TDM
// 2-D TDM load: tile_h rows of tile_w bf16 elements, tensor row stride stride0
// (elements), into LDS at lds_off (packed row-major). D# per CDNA5 ISA ch.8.
__device__ __forceinline__ void tdm_load_2d(unsigned lds_off, const void* gaddr,
                                            unsigned tile_w, unsigned tile_h,
                                            unsigned tensor_w, unsigned tensor_h,
                                            unsigned stride0) {
  unsigned long long ga = (unsigned long long)gaddr;
  v4u g0;
  g0[0] = 1u;                                                // count=1
  g0[1] = lds_off;                                           // lds_addr
  g0[2] = (unsigned)(ga & 0xffffffffu);                      // global_addr lo
  g0[3] = (unsigned)((ga >> 32) & 0x1ffffffu) | (2u << 30);  // addr hi | type=2
  v8i g1;
  g1[0] = (int)(1u << 16);                                   // data_size=1 (2B)
  g1[1] = (int)((tensor_w & 0xffffu) << 16);
  g1[2] = (int)((tensor_w >> 16) & 0xffffu) | (int)((tensor_h & 0xffffu) << 16);
  g1[3] = (int)((tensor_h >> 16) & 0xffffu) | (int)(tile_w << 16);
  g1[4] = (int)(tile_h & 0xffffu);
  g1[5] = (int)stride0;
  g1[6] = 0;
  g1[7] = 0;
  v4i z = {0, 0, 0, 0};
#if __clang_major__ >= 23
  v8i z8 = {0, 0, 0, 0, 0, 0, 0, 0};
  __builtin_amdgcn_tensor_load_to_lds(g0, g1, z, z, z8, 0);
#else
  __builtin_amdgcn_tensor_load_to_lds(g0, g1, z, z, 0);
#endif
}
#endif

// ---------------------------------------------------------------- fp32 -> bf16
__global__ void cvt_f32_bf16(const float* __restrict__ s, bf16_t* __restrict__ d, long n) {
  long i = (long)blockIdx.x * blockDim.x + threadIdx.x;
  long st = (long)gridDim.x * blockDim.x;
  for (; i < n; i += st) d[i] = (bf16_t)s[i];
}

// ---------------------------------------------------------------- transpose + cvt
// out[z][c][r] = (bf16) in[z_in][r][c];  in rows R x cols C, row stride ldi.
// per-z input offset = (z/zin_cnt)*in_zs_o + (z%zin_cnt)*in_zs_i.
__global__ __launch_bounds__(256) void transpose_cvt(
    const float* __restrict__ in, bf16_t* __restrict__ out,
    int R, int C, int ldi,
    long in_zs_o, long in_zs_i, int zin_cnt, long out_zs) {
  __shared__ float t[32][33];
  const int z = blockIdx.z;
  const float* ip = in + (long)(z / zin_cnt) * in_zs_o + (long)(z % zin_cnt) * in_zs_i;
  bf16_t* op = out + (long)z * out_zs;
  const int r0 = blockIdx.y * 32, c0 = blockIdx.x * 32;
  const int tr = threadIdx.x >> 3;
  const int tc = (threadIdx.x & 7) * 4;
  float4 v = *(const float4*)(ip + (size_t)(r0 + tr) * ldi + c0 + tc);
  t[tr][tc + 0] = v.x; t[tr][tc + 1] = v.y; t[tr][tc + 2] = v.z; t[tr][tc + 3] = v.w;
  __syncthreads();
  bf16_t* o4 = op + (size_t)(c0 + tr) * R + r0 + tc;
  o4[0] = (bf16_t)t[tc + 0][tr];
  o4[1] = (bf16_t)t[tc + 1][tr];
  o4[2] = (bf16_t)t[tc + 2][tr];
  o4[3] = (bf16_t)t[tc + 3][tr];
}

// ---------------------------------------------------------------- bf16 WMMA GEMM
// C[M,N] (f32) = A[M,K] (bf16 rm) * Btg^T, with Btg = B transposed [N,K] bf16.
// 64x64 tile / workgroup, 4 waves each 32x32, K-step 32.
// Both A and B tiles arrive via TDM, double buffered.
__global__ __launch_bounds__(128) void gemm_bf16_wmma(
    const bf16_t* __restrict__ A, const bf16_t* __restrict__ Btg,
    float* __restrict__ C, int M, int N, int K) {
  __shared__ __align__(16) bf16_t As[2][64 * 32];
  __shared__ __align__(16) bf16_t Bs[2][64 * 32];
  __shared__ __align__(16) float Cs[64 * 64];

  const int tid  = threadIdx.x;
  const int wave = tid >> 5;
  const int lane = tid & 31;
  const int hsel = lane >> 4;
  const int lr   = lane & 15;

  const int m0 = blockIdx.y * 64;
  const int n0 = blockIdx.x * 64;
  const int wm = (wave >> 1) * 32;
  const int wn = (wave & 1) * 32;
  const int nk = K >> 5;

  v8f acc[2][2] = {};

#if !HAVE_TDM
  auto stageA = [&](int buf, int k0) {
    int row = tid >> 1;
    int off = (tid & 1) * 16;
    const uint4* src = (const uint4*)(A + (size_t)(m0 + row) * K + k0 + off);
    uint4* dst = (uint4*)(&As[buf][row * 32 + off]);
    dst[0] = src[0];
    dst[1] = src[1];
  };
  auto stageB = [&](int buf, int k0) {
    int row = tid >> 1;
    int off = (tid & 1) * 16;
    const uint4* src = (const uint4*)(Btg + (size_t)(n0 + row) * K + k0 + off);
    uint4* dst = (uint4*)(&Bs[buf][row * 32 + off]);
    dst[0] = src[0];
    dst[1] = src[1];
  };
#endif

  // prologue: stage k-step 0
#if HAVE_TDM
  if (wave == 0) {
    tdm_load_2d((unsigned)(size_t)&As[0][0], A + (size_t)m0 * K, 32u, 64u,
                (unsigned)K, (unsigned)M, (unsigned)K);
    tdm_load_2d((unsigned)(size_t)&Bs[0][0], Btg + (size_t)n0 * K, 32u, 64u,
                (unsigned)K, (unsigned)N, (unsigned)K);
  }
#else
  stageA(0, 0);
  stageB(0, 0);
#endif

  for (int i = 0; i < nk; ++i) {
    const int buf = i & 1;
#if HAVE_TDM
    if (wave == 0) __builtin_amdgcn_s_wait_tensorcnt(0);
#endif
    __syncthreads();  // tile i ready; everyone done with tile i-1
    if (i + 1 < nk) {
#if HAVE_TDM
      if (wave == 0) {
        tdm_load_2d((unsigned)(size_t)&As[buf ^ 1][0],
                    A + (size_t)m0 * K + (size_t)(i + 1) * 32, 32u, 64u,
                    (unsigned)K, (unsigned)M, (unsigned)K);
        tdm_load_2d((unsigned)(size_t)&Bs[buf ^ 1][0],
                    Btg + (size_t)n0 * K + (size_t)(i + 1) * 32, 32u, 64u,
                    (unsigned)K, (unsigned)N, (unsigned)K);
      }
#else
      stageA(buf ^ 1, (i + 1) * 32);
      stageB(buf ^ 1, (i + 1) * 32);
#endif
    }

    v16bf af[2], bfr[2];
#pragma unroll
    for (int ii = 0; ii < 2; ++ii) {
      const bf16_t* ar = &As[buf][(wm + ii * 16 + lr) * 32];
#pragma unroll
      for (int p = 0; p < 8; ++p) {
        int kk = kmap(p, hsel);
        af[ii][2 * p]     = ar[kk];
        af[ii][2 * p + 1] = ar[kk + 1];
      }
    }
#pragma unroll
    for (int j = 0; j < 2; ++j) {
      const bf16_t* bc = &Bs[buf][(wn + j * 16 + lr) * 32];
#pragma unroll
      for (int p = 0; p < 8; ++p) {
        int kk = kmap(p, hsel);
        bfr[j][2 * p]     = bc[kk];
        bfr[j][2 * p + 1] = bc[kk + 1];
      }
    }
#pragma unroll
    for (int ii = 0; ii < 2; ++ii)
#pragma unroll
      for (int j = 0; j < 2; ++j)
        acc[ii][j] = __builtin_amdgcn_wmma_f32_16x16x32_bf16(
            false, af[ii], false, bfr[j], (short)0, acc[ii][j], false, false);
  }

  // epilogue: stage C tile in LDS, write out coalesced float4
#pragma unroll
  for (int ii = 0; ii < 2; ++ii)
#pragma unroll
    for (int j = 0; j < 2; ++j)
#pragma unroll
      for (int g = 0; g < 8; ++g)
        Cs[(wm + ii * 16 + hsel * 8 + g) * 64 + wn + j * 16 + lr] = acc[ii][j][g];
  __syncthreads();
  float* Cb = C + (size_t)m0 * N + n0;
#pragma unroll
  for (int rep = 0; rep < 8; ++rep) {
    int u = rep * 128 + tid;
    int row = u >> 4, q = u & 15;
    int off = row * N + q * 4;
    *(float4*)(Cb + off) = *(const float4*)&Cs[row * 64 + q * 4];
  }
}

// ---------------------------------------------------------------- q prep
__global__ __launch_bounds__(256) void prep_q(const float* __restrict__ q32,
                                              float* __restrict__ ql,
                                              bf16_t* __restrict__ qh) {
  int rid  = blockIdx.x * 8 + (threadIdx.x >> 5);
  int lane = threadIdx.x & 31;
  const float4* src = (const float4*)(q32 + (size_t)rid * 128);
  float4 x = src[lane];
  int h = rid & 15;
  long bn = rid >> 4;
  long b = bn >> 11, n = bn & 2047;
  size_t dst = (((size_t)(b * 16 + h)) * 2048 + n) * 128;

  float m = fmaxf(fmaxf(x.x, x.y), fmaxf(x.z, x.w));
  for (int mm = 16; mm >= 1; mm >>= 1) m = fmaxf(m, __shfl_xor(m, mm, 32));
  float e0 = expf(x.x - m), e1 = expf(x.y - m), e2 = expf(x.z - m), e3 = expf(x.w - m);
  float s = e0 + e1 + e2 + e3;
  for (int mm = 16; mm >= 1; mm >>= 1) s += __shfl_xor(s, mm, 32);
  float inv = 1.0f / s;
  float4 r = {e0 * inv, e1 * inv, e2 * inv, e3 * inv};
  ((float4*)(ql + dst))[lane] = r;
  bf16_t* qd = qh + dst + lane * 4;
  qd[0] = (bf16_t)x.x; qd[1] = (bf16_t)x.y; qd[2] = (bf16_t)x.z; qd[3] = (bf16_t)x.w;
}

// ---------------------------------------------------------------- k prep
__global__ __launch_bounds__(256) void prep_kv(const float* __restrict__ k32,
                                               bf16_t* __restrict__ kh,
                                               float* __restrict__ kl,
                                               float* __restrict__ bet) {
  int rid  = blockIdx.x * 8 + (threadIdx.x >> 5);
  int lane = threadIdx.x & 31;
  int kv = rid & 3;
  long bn = rid >> 2;
  long b = bn >> 11, n = bn & 2047;
  const float4* ks = (const float4*)(k32 + bn * 512 + kv * 128);
  float4 x = ks[lane];
  size_t dst = (((size_t)(b * 4 + kv)) * 2048 + n) * 128;

  float m = fmaxf(fmaxf(x.x, x.y), fmaxf(x.z, x.w));
  for (int mm = 16; mm >= 1; mm >>= 1) m = fmaxf(m, __shfl_xor(m, mm, 32));
  float e0 = expf(x.x - m), e1 = expf(x.y - m), e2 = expf(x.z - m), e3 = expf(x.w - m);
  float s = e0 + e1 + e2 + e3;
  for (int mm = 16; mm >= 1; mm >>= 1) s += __shfl_xor(s, mm, 32);
  float inv = 1.0f / s;
  float4 r = {e0 * inv, e1 * inv, e2 * inv, e3 * inv};
  ((float4*)(kl + dst))[lane] = r;

  float xa[4] = {x.x, x.y, x.z, x.w};
  bf16_t* kd = kh + dst + lane * 4;
  float* bd = bet + dst + lane * 4;
#pragma unroll
  for (int i = 0; i < 4; ++i) {
    kd[i] = (bf16_t)xa[i];
    float ls = fminf(xa[i], 0.f) - log1pf(expf(-fabsf(xa[i])));
    bd[i] = expf(ls * 0.0625f);
  }
}

// ---------------------------------------------------------------- causal flash attention
// qh:[B,H,N,128] kh:[B,KVH,N,128] vt:[B,KVH,128,N]; obase:[B,N,H*128]
__global__ __launch_bounds__(128) void attn_base(const bf16_t* __restrict__ qh,
                                                 const bf16_t* __restrict__ kh,
                                                 const bf16_t* __restrict__ vt,
                                                 float* __restrict__ obase) {
  __shared__ __align__(16) bf16_t Qs[64 * 128];
  __shared__ __align__(16) bf16_t Ks[64 * 128];
  __shared__ __align__(16) bf16_t Vt[128 * 64];  // (col d, key)
  __shared__ __align__(16) bf16_t Ps[4][16 * 64];

  const int tid  = threadIdx.x;
  const int wave = tid >> 5;
  const int lane = tid & 31;
  const int hsel = lane >> 4;
  const int lr   = lane & 15;

  const int bh = blockIdx.x;
  const int b = bh >> 4, h = bh & 15, kv = h >> 2;
  const int qi0 = blockIdx.y * 64;

  const bf16_t* Qg = qh + ((size_t)(b * 16 + h) * 2048) * 128;
  const bf16_t* Kg = kh + ((size_t)(b * 4 + kv) * 2048) * 128;
  const bf16_t* Vg = vt + ((size_t)(b * 4 + kv) * 128) * 2048;

  // Q tile (contiguous 64x128)
#if HAVE_TDM
  if (wave == 0)
    tdm_load_2d((unsigned)(size_t)&Qs[0], Qg + (size_t)qi0 * 128, 8192u, 1u,
                8192u, 1u, 8192u);
#else
  {
    const uint4* s = (const uint4*)(Qg + (size_t)qi0 * 128);
    uint4* d = (uint4*)Qs;
#pragma unroll
    for (int u = 0; u < 8; ++u) d[tid * 8 + u] = s[tid * 8 + u];
  }
#endif

  float mrow[8], lrow[8];
  v8f o[8] = {};
#pragma unroll
  for (int g = 0; g < 8; ++g) { mrow[g] = -3.0e38f; lrow[g] = 0.f; }

  const float scale = 0.08838834764831845f;  // 1/sqrt(128)

  for (int j0 = 0; j0 <= qi0; j0 += 64) {
    __syncthreads();  // everyone done with previous K/V tiles
#if HAVE_TDM
    if (wave == 0) {
      tdm_load_2d((unsigned)(size_t)&Ks[0], Kg + (size_t)j0 * 128, 8192u, 1u,
                  8192u, 1u, 8192u);
      tdm_load_2d((unsigned)(size_t)&Vt[0], Vg + j0, 64u, 128u,
                  2048u, 128u, 2048u);
      __builtin_amdgcn_s_wait_tensorcnt(0);
    }
#else
    {
      const uint4* s1 = (const uint4*)(Kg + (size_t)j0 * 128);
      uint4* d1 = (uint4*)Ks;
#pragma unroll
      for (int u = 0; u < 8; ++u) d1[tid * 8 + u] = s1[tid * 8 + u];
      const uint4* s2 = (const uint4*)(Vg + (size_t)tid * 2048 + j0);
      uint4* d2 = (uint4*)(&Vt[tid * 64]);
#pragma unroll
      for (int u = 0; u < 8; ++u) d2[u] = s2[u];
    }
#endif
    __syncthreads();

    // S = Q * K^T  (16x64 per wave)
    v8f sc[4] = {};
#pragma unroll
    for (int kk0 = 0; kk0 < 128; kk0 += 32) {
      v16bf a;
#pragma unroll
      for (int p = 0; p < 8; ++p) {
        int kk = kmap(p, hsel);
        a[2 * p]     = Qs[(wave * 16 + lr) * 128 + kk0 + kk];
        a[2 * p + 1] = Qs[(wave * 16 + lr) * 128 + kk0 + kk + 1];
      }
#pragma unroll
      for (int jt = 0; jt < 4; ++jt) {
        v16bf bb;
#pragma unroll
        for (int p = 0; p < 8; ++p) {
          int kk = kmap(p, hsel);
          bb[2 * p]     = Ks[(jt * 16 + lr) * 128 + kk0 + kk];
          bb[2 * p + 1] = Ks[(jt * 16 + lr) * 128 + kk0 + kk + 1];
        }
        sc[jt] = __builtin_amdgcn_wmma_f32_16x16x32_bf16(false, a, false, bb, (short)0, sc[jt], false, false);
      }
    }

    // online softmax (row stats shared by the 16-lane half)
    const bool diag = (j0 == qi0);
#pragma unroll
    for (int g = 0; g < 8; ++g) {
      int mloc = hsel * 8 + g;
      int qrow = qi0 + wave * 16 + mloc;
      float rmax = -3.0e38f;
#pragma unroll
      for (int jt = 0; jt < 4; ++jt) {
        float v = sc[jt][g] * scale;
        if (diag && (j0 + jt * 16 + lr) > qrow) v = -3.0e38f;
        sc[jt][g] = v;
        rmax = fmaxf(rmax, v);
      }
      rmax = red16_max(rmax);
      float mnew = fmaxf(mrow[g], rmax);
      float scl = expf(mrow[g] - mnew);
      mrow[g] = mnew;
      float rsum = 0.f;
#pragma unroll
      for (int jt = 0; jt < 4; ++jt) {
        float p = expf(sc[jt][g] - mnew);
        sc[jt][g] = p;
        rsum += p;
      }
      rsum = red16_sum(rsum);
      lrow[g] = lrow[g] * scl + rsum;
#pragma unroll
      for (int t = 0; t < 8; ++t) o[t][g] *= scl;
#pragma unroll
      for (int jt = 0; jt < 4; ++jt)
        Ps[wave][mloc * 64 + jt * 16 + lr] = (bf16_t)sc[jt][g];
    }

    asm volatile("s_wait_dscnt 0" ::: "memory");

    // O += P * V
#pragma unroll
    for (int kk0 = 0; kk0 < 64; kk0 += 32) {
      v16bf a;
#pragma unroll
      for (int p = 0; p < 8; ++p) {
        int kk = kmap(p, hsel);
        a[2 * p]     = Ps[wave][lr * 64 + kk0 + kk];
        a[2 * p + 1] = Ps[wave][lr * 64 + kk0 + kk + 1];
      }
#pragma unroll
      for (int ot = 0; ot < 8; ++ot) {
        const bf16_t* vc = &Vt[(ot * 16 + lr) * 64];
        v16bf bb;
#pragma unroll
        for (int p = 0; p < 8; ++p) {
          int kk = kmap(p, hsel);
          bb[2 * p]     = vc[kk0 + kk];
          bb[2 * p + 1] = vc[kk0 + kk + 1];
        }
        o[ot] = __builtin_amdgcn_wmma_f32_16x16x32_bf16(false, a, false, bb, (short)0, o[ot], false, false);
      }
    }
  }

#pragma unroll
  for (int g = 0; g < 8; ++g) {
    int mloc = hsel * 8 + g;
    int qrow = qi0 + wave * 16 + mloc;
    float inv = 1.0f / lrow[g];
    size_t base = ((size_t)b * 2048 + qrow) * 2048 + h * 128;
#pragma unroll
    for (int ot = 0; ot < 8; ++ot) obase[base + ot * 16 + lr] = o[ot][g] * inv;
  }
}

// ---------------------------------------------------------------- delta-rule scan
// One wave32 per block (free barriers, in-order same-wave LDS). Lane =
// (column e in 0..7, d-group g in 0..3); each lane owns 32 state elements
// S[d in 32g..32g+31][e] in VGPRs. pred/o reduced across the 4 d-groups with
// __shfl_xor(8/16). 512 blocks = 32 (b,h) x 16 column-blocks.
__global__ __launch_bounds__(32) void delta_scan(const float* __restrict__ ql,
                                                 const float* __restrict__ kl,
                                                 const float* __restrict__ bet,
                                                 const float* __restrict__ v32,
                                                 float* __restrict__ olin) {
  __shared__ __align__(16) float sk[128], skb[128], sq[128];
  const int lane = threadIdx.x;
  const int g = lane >> 3;    // d-group
  const int el = lane & 7;    // local column
  const int blk = blockIdx.x;
  const int bh = blk >> 4;
  const int eb = blk & 15;    // column block (8 columns each)
  const int b = bh >> 4, h = bh & 15, kv = h >> 2;
  const int e = eb * 8 + el;

  const float* qbase = ql + ((size_t)(b * 16 + h) * 2048) * 128;
  const float* kbase = kl + ((size_t)(b * 4 + kv) * 2048) * 128;
  const float* bbase = bet + ((size_t)(b * 4 + kv) * 2048) * 128;
  const float* vbase = v32 + (size_t)b * 2048 * 512 + kv * 128 + e;
  float* ob = olin + (size_t)b * 2048 * 2048 + h * 128 + e;

  float S[32];
#pragma unroll
  for (int d = 0; d < 32; ++d) S[d] = 0.f;

  const int d0 = g * 32;
  for (int t = 0; t < 2048; ++t) {
    // stage shared row vectors (wave-internal; LDS in order, no barrier needed)
    float4 kx = *(const float4*)(kbase + (size_t)t * 128 + lane * 4);
    float4 qx = *(const float4*)(qbase + (size_t)t * 128 + lane * 4);
    float4 bx = *(const float4*)(bbase + (size_t)t * 128 + lane * 4);
    *(float4*)&sk[lane * 4] = kx;
    *(float4*)&sq[lane * 4] = qx;
    float4 kb4 = {kx.x * bx.x, kx.y * bx.y, kx.z * bx.z, kx.w * bx.w};
    *(float4*)&skb[lane * 4] = kb4;
    float vv = vbase[(size_t)t * 512];

    float pred = 0.f;
#pragma unroll
    for (int i = 0; i < 32; ++i) pred += sk[d0 + i] * S[i];
    pred += __shfl_xor(pred, 8, 32);
    pred += __shfl_xor(pred, 16, 32);
    float delta = vv - pred;

    float ov = 0.f;
#pragma unroll
    for (int i = 0; i < 32; ++i) {
      S[i] += skb[d0 + i] * delta;
      ov += sq[d0 + i] * S[i];
    }
    ov += __shfl_xor(ov, 8, 32);
    ov += __shfl_xor(ov, 16, 32);
    if (g == 0) ob[(size_t)t * 2048] = ov;
  }
}

// ---------------------------------------------------------------- blend (MAG = 0.5)
__global__ void blend_kernel(const float* __restrict__ a, const float* __restrict__ b,
                             bf16_t* __restrict__ d, long n) {
  long i = (long)blockIdx.x * blockDim.x + threadIdx.x;
  long st = (long)gridDim.x * blockDim.x;
  for (; i < n; i += st) d[i] = (bf16_t)(0.5f * (a[i] + b[i]));
}

// ---------------------------------------------------------------- launch
extern "C" void kernel_launch(void* const* d_in, const int* in_sizes, int n_in,
                              void* d_out, int out_size, void* d_ws, size_t ws_size,
                              hipStream_t stream) {
  const float* hs = (const float*)d_in[0];
  const float* wq = (const float*)d_in[1];
  const float* wk = (const float*)d_in[2];
  const float* wv = (const float*)d_in[3];
  const float* wo = (const float*)d_in[4];
  float* out = (float*)d_out;

  const size_t M = 4096;  // B*N

  char* cur = (char*)d_ws;
  auto alloc = [&](size_t bytes) -> void* {
    void* r = (void*)cur;
    cur += (bytes + 255) & ~(size_t)255;
    return r;
  };

  bf16_t* hs_b  = (bf16_t*)alloc(M * 2048 * 2);
  bf16_t* wqt   = (bf16_t*)alloc((size_t)2048 * 2048 * 2);  // [N,K]
  bf16_t* wkt   = (bf16_t*)alloc((size_t)512 * 2048 * 2);
  bf16_t* wvt   = (bf16_t*)alloc((size_t)512 * 2048 * 2);
  bf16_t* wot   = (bf16_t*)alloc((size_t)2048 * 2048 * 2);
  float*  q32   = (float*)alloc(M * 2048 * 4);
  float*  k32   = (float*)alloc(M * 512 * 4);
  float*  v32   = (float*)alloc(M * 512 * 4);
  bf16_t* qh_b  = (bf16_t*)alloc(M * 2048 * 2);
  bf16_t* kh_b  = (bf16_t*)alloc(M * 512 * 2);
  bf16_t* vt_b  = (bf16_t*)alloc(M * 512 * 2);              // [B,KVH,128,N]
  float*  ql32  = (float*)alloc(M * 2048 * 4);
  float*  kl32  = (float*)alloc(M * 512 * 4);
  float*  bet32 = (float*)alloc(M * 512 * 4);
  float*  olin  = (float*)alloc(M * 2048 * 4);
  float*  obase = (float*)alloc(M * 2048 * 4);
  bf16_t* bl_b  = (bf16_t*)alloc(M * 2048 * 2);

  cvt_f32_bf16<<<2048, 256, 0, stream>>>(hs, hs_b, (long)M * 2048);
  transpose_cvt<<<dim3(64, 64, 1), 256, 0, stream>>>(wq, wqt, 2048, 2048, 2048, 0, 0, 1, 0);
  transpose_cvt<<<dim3(16, 64, 1), 256, 0, stream>>>(wk, wkt, 2048, 512, 512, 0, 0, 1, 0);
  transpose_cvt<<<dim3(16, 64, 1), 256, 0, stream>>>(wv, wvt, 2048, 512, 512, 0, 0, 1, 0);
  transpose_cvt<<<dim3(64, 64, 1), 256, 0, stream>>>(wo, wot, 2048, 2048, 2048, 0, 0, 1, 0);

  gemm_bf16_wmma<<<dim3(2048 / 64, M / 64), 128, 0, stream>>>(hs_b, wqt, q32, (int)M, 2048, 2048);
  gemm_bf16_wmma<<<dim3(512 / 64,  M / 64), 128, 0, stream>>>(hs_b, wkt, k32, (int)M, 512, 2048);
  gemm_bf16_wmma<<<dim3(512 / 64,  M / 64), 128, 0, stream>>>(hs_b, wvt, v32, (int)M, 512, 2048);

  prep_q<<<(2 * 2048 * 16) / 8, 256, 0, stream>>>(q32, ql32, qh_b);
  prep_kv<<<(2 * 2048 * 4) / 8, 256, 0, stream>>>(k32, kh_b, kl32, bet32);
  // v (f32 [B,N,KVH*128]) -> per-head transposed bf16 [B,KVH,128,N]
  transpose_cvt<<<dim3(4, 64, 8), 256, 0, stream>>>(v32, vt_b, 2048, 128, 512,
                                                    (long)2048 * 512, 128, 4,
                                                    (long)128 * 2048);

  attn_base<<<dim3(32, 32), 128, 0, stream>>>(qh_b, kh_b, vt_b, obase);
  delta_scan<<<512, 32, 0, stream>>>(ql32, kl32, bet32, v32, olin);
  blend_kernel<<<2048, 256, 0, stream>>>(olin, obase, bl_b, (long)M * 2048);

  gemm_bf16_wmma<<<dim3(2048 / 64, M / 64), 128, 0, stream>>>(bl_b, wot, out, (int)M, 2048, 2048);
}
